// CausalAttention_29463475651079
// MI455X (gfx1250) — matrix operand order repaired and checked by reading, compiled, not run on hardware
//
#include <hip/hip_runtime.h>
#include <hip/hip_bf16.h>
#include <math.h>

#define DI __device__ __forceinline__

typedef __attribute__((ext_vector_type(16))) __bf16 v16bf;
typedef __attribute__((ext_vector_type(8)))  float  v8f;

union Frag16 { unsigned int u[8]; v16bf v; };

constexpr int BB = 2, S = 2048, C = 1024, NH = 16, HD = 64;
constexpr int M_ROWS = BB * S;        // 4096
constexpr int N_QKV  = 3 * C;         // 3072

// workspace offsets in ushort (bf16) elements
constexpr size_t XB_OFF = 0;
constexpr size_t WQ_OFF = XB_OFF + (size_t)M_ROWS * C;        // x bf16
constexpr size_t WO_OFF = WQ_OFF + (size_t)N_QKV * C;         // Wqkv bf16
constexpr size_t Q_OFF  = WO_OFF + (size_t)C * C;             // Wo bf16
constexpr size_t K_OFF  = Q_OFF  + (size_t)BB * NH * S * HD;  // Q [b,h,s,d]
constexpr size_t VT_OFF = K_OFF  + (size_t)BB * NH * S * HD;  // K [b,h,s,d]
constexpr size_t OC_OFF = VT_OFF + (size_t)BB * NH * S * HD;  // V [b,h,d,s]
// OC: attention output [b,s,c] bf16; ~48MB total workspace.

DI unsigned short f2bf(float f) {
  union { float f; unsigned int u; } c; c.f = f;
  unsigned int r = c.u + 0x7FFFu + ((c.u >> 16) & 1u);   // RNE
  return (unsigned short)(r >> 16);
}
DI unsigned int pack2(float lo, float hi) {
  return (unsigned int)f2bf(lo) | ((unsigned int)f2bf(hi) << 16);
}
DI v8f zero8() { v8f z = {0.f,0.f,0.f,0.f,0.f,0.f,0.f,0.f}; return z; }
// A-operand (16x32 bf16) K-pair start for vgpr v, lane-half hh (contiguous: k = 16*hh + 2*v)
DI int a_k0(int v, int hh) { return 2*v + (hh ? 16 : 0); }

// ---------------- fp32 -> bf16 conversion ----------------
__global__ void cvt_bf16(const float* __restrict__ src,
                         unsigned short* __restrict__ dst, int n) {
  int i = blockIdx.x * blockDim.x + threadIdx.x;
  if (i < n) dst[i] = f2bf(src[i]);
}

// ---------------- QKV projection GEMM ----------------
// A = x_bf16 [4096,1024], B = Wqkv_bf16 [3072,1024] (n-major, k contiguous)
// wave tile 64x64 (4x4 frags, 16 WMMAs / 8 frag loads per K=32 step)
// block = 4 waves stacked in M -> 256x64 per block
__global__ void __launch_bounds__(128) qkv_gemm(
    const unsigned short* __restrict__ Ab,
    const unsigned short* __restrict__ Wb,
    const float* __restrict__ bias,
    unsigned short* __restrict__ Qb,
    unsigned short* __restrict__ Kb,
    unsigned short* __restrict__ VT) {
  int lane = threadIdx.x & 31, wid = threadIdx.x >> 5;
  int hh = lane >> 4, ln = lane & 15;
  int mbase = blockIdx.x * 256 + wid * 64;
  int nbase = blockIdx.y * 64;

  v8f acc[4][4];
#pragma unroll
  for (int mi = 0; mi < 4; ++mi)
#pragma unroll
    for (int ni = 0; ni < 4; ++ni) acc[mi][ni] = zero8();

  for (int kt = 0; kt < C; kt += 32) {
    Frag16 af[4], bf[4];
#pragma unroll
    for (int mi = 0; mi < 4; ++mi) {
      const unsigned short* arow = Ab + (size_t)(mbase + mi*16 + ln) * C + kt;
      __builtin_prefetch(arow + 32, 0, 1);          // next K tile
#pragma unroll
      for (int v = 0; v < 8; ++v)
        af[mi].u[v] = *(const unsigned int*)(arow + a_k0(v, hh));
    }
#pragma unroll
    for (int ni = 0; ni < 4; ++ni) {
      const unsigned short* brow = Wb + (size_t)(nbase + ni*16 + ln) * C + kt + 16*hh;
      __builtin_prefetch(brow + 32, 0, 1);          // next K tile
#pragma unroll
      for (int v = 0; v < 8; ++v)
        bf[ni].u[v] = *(const unsigned int*)(brow + 2*v);
    }
#pragma unroll
    for (int mi = 0; mi < 4; ++mi)
#pragma unroll
      for (int ni = 0; ni < 4; ++ni)
        acc[mi][ni] = __builtin_amdgcn_wmma_f32_16x16x32_bf16(
            false, af[mi].v, false, bf[ni].v, (short)0, acc[mi][ni], false, false);
  }

#pragma unroll
  for (int ni = 0; ni < 4; ++ni) {
    int ng = nbase + ni*16 + ln;
    float bi = bias[ng];
    int which = ng >> 10;            // 0=Q 1=K 2=V
    int cm = ng & 1023;
    int head = cm >> 6, feat = cm & 63;
#pragma unroll
    for (int mi = 0; mi < 4; ++mi) {
#pragma unroll
      for (int v = 0; v < 8; ++v) {
        int mg = mbase + mi*16 + v + 8*hh;       // D layout: row = v + 8*(lane/16)
        int bb = mg >> 11, ss = mg & 2047;
        unsigned short val = f2bf(acc[mi][ni][v] + bi);
        size_t bhh = (size_t)(bb * NH + head);
        if (which == 0)      Qb[(bhh * S + ss) * HD + feat] = val;
        else if (which == 1) Kb[(bhh * S + ss) * HD + feat] = val;
        else                 VT[(bhh * HD + feat) * S + ss] = val;  // V transposed
      }
    }
  }
}

// ---------------- Flash attention ----------------
// one wave per 16-row Q tile; 32-key chunks; online softmax
__global__ void __launch_bounds__(128) flash_attn(
    const unsigned short* __restrict__ Qb,
    const unsigned short* __restrict__ Kb,
    const unsigned short* __restrict__ VT,
    const unsigned char*  __restrict__ padmask,
    unsigned short* __restrict__ Oc) {
  __shared__ float tile[4][16][32];
  int lane = threadIdx.x & 31, wid = threadIdx.x >> 5;
  int g = blockIdx.x * 4 + wid;          // 4096 waves total
  int qtile = g & 127, bh = g >> 7;
  int bidx = bh >> 4;
  int q0 = qtile * 16;
  int hh = lane >> 4, ln = lane & 15;

  Frag16 qf[2];
  {
    const unsigned short* qrow = Qb + ((size_t)bh * S + q0 + ln) * HD;
#pragma unroll
    for (int fi = 0; fi < 2; ++fi)
#pragma unroll
      for (int v = 0; v < 8; ++v)
        qf[fi].u[v] = *(const unsigned int*)(qrow + a_k0(v, hh) + 32*fi);
  }

  float Mrow[8], Lrow[8];
  v8f oacc[4];
#pragma unroll
  for (int v = 0; v < 8; ++v) { Mrow[v] = -3.0e38f; Lrow[v] = 0.f; }
#pragma unroll
  for (int f = 0; f < 4; ++f) oacc[f] = zero8();

  const float scale = 0.125f;           // 1/sqrt(64)
  int nkt = (q0 + 16 + 31) >> 5;        // causal: chunks with kb <= q0+15

  for (int kc = 0; kc < nkt; ++kc) {
    int kb = kc << 5;
    // prefetch next chunk's K rows while this chunk computes
    if (kc + 1 < nkt) {
      const unsigned short* knext = Kb + ((size_t)bh * S + kb + 32 + lane) * HD;
      __builtin_prefetch(knext, 0, 1);
    }
    v8f s[2];
#pragma unroll
    for (int nt = 0; nt < 2; ++nt) {
      Frag16 kf0, kf1;
      const unsigned short* krow =
          Kb + ((size_t)bh * S + kb + nt*16 + ln) * HD + 16*hh;
#pragma unroll
      for (int v = 0; v < 8; ++v) kf0.u[v] = *(const unsigned int*)(krow + 2*v);
#pragma unroll
      for (int v = 0; v < 8; ++v) kf1.u[v] = *(const unsigned int*)(krow + 32 + 2*v);
      v8f sc = zero8();
      sc = __builtin_amdgcn_wmma_f32_16x16x32_bf16(false, qf[0].v, false, kf0.v,
                                                   (short)0, sc, false, false);
      sc = __builtin_amdgcn_wmma_f32_16x16x32_bf16(false, qf[1].v, false, kf1.v,
                                                   (short)0, sc, false, false);
      s[nt] = sc;
    }

    int col0 = kb + ln, col1 = col0 + 16;
    bool pm0 = padmask[(size_t)bidx * S + col0] != 0;
    bool pm1 = padmask[(size_t)bidx * S + col1] != 0;

#pragma unroll
    for (int v = 0; v < 8; ++v) {
      int row = q0 + v + 8*hh;
      float a  = s[0][v] * scale;
      float b2 = s[1][v] * scale;
      if (col0 > row || pm0) a  = -3.0e38f;
      if (col1 > row || pm1) b2 = -3.0e38f;
      float mx = fmaxf(a, b2);
      mx = fmaxf(mx, __shfl_xor(mx, 1, 32));
      mx = fmaxf(mx, __shfl_xor(mx, 2, 32));
      mx = fmaxf(mx, __shfl_xor(mx, 4, 32));
      mx = fmaxf(mx, __shfl_xor(mx, 8, 32));
      float Mn = fmaxf(Mrow[v], mx);
      float alpha = __expf(Mrow[v] - Mn);
      float e0 = __expf(a - Mn), e1 = __expf(b2 - Mn);
      float rs = e0 + e1;
      rs += __shfl_xor(rs, 1, 32);
      rs += __shfl_xor(rs, 2, 32);
      rs += __shfl_xor(rs, 4, 32);
      rs += __shfl_xor(rs, 8, 32);
      Lrow[v] = Lrow[v] * alpha + rs;
      Mrow[v] = Mn;
#pragma unroll
      for (int f = 0; f < 4; ++f) oacc[f][v] *= alpha;
      tile[wid][v + 8*hh][ln]      = e0;     // re-layout D->A via LDS
      tile[wid][v + 8*hh][ln + 16] = e1;
    }

    asm volatile("s_wait_dscnt 0" ::: "memory");

    Frag16 pf;
#pragma unroll
    for (int v = 0; v < 8; ++v) {
      int k0 = a_k0(v, hh);
      pf.u[v] = pack2(tile[wid][ln][k0], tile[wid][ln][k0 + 1]);
    }

#pragma unroll
    for (int f = 0; f < 4; ++f) {
      Frag16 vf;   // V transposed [d,s]: keys contiguous -> dword loads
      const unsigned short* vrow =
          VT + ((size_t)bh * HD + f*16 + ln) * S + kb + 16*hh;
      if (kc + 1 < nkt) __builtin_prefetch(vrow + 32, 0, 1);
#pragma unroll
      for (int v = 0; v < 8; ++v) vf.u[v] = *(const unsigned int*)(vrow + 2*v);
      oacc[f] = __builtin_amdgcn_wmma_f32_16x16x32_bf16(
          false, pf.v, false, vf.v, (short)0, oacc[f], false, false);
    }
  }

#pragma unroll
  for (int v = 0; v < 8; ++v) {
    float inv = 1.0f / Lrow[v];
    int row = q0 + v + 8*hh;
    size_t base = ((size_t)bidx * S + row) * C + (size_t)(bh & 15) * HD;
#pragma unroll
    for (int f = 0; f < 4; ++f)
      Oc[base + f*16 + ln] = f2bf(oacc[f][v] * inv);
  }
}

// ---------------- Output projection GEMM ----------------
__global__ void __launch_bounds__(128) proj_gemm(
    const unsigned short* __restrict__ Ab,
    const unsigned short* __restrict__ Wb,
    const float* __restrict__ bias,
    float* __restrict__ out) {
  int lane = threadIdx.x & 31, wid = threadIdx.x >> 5;
  int hh = lane >> 4, ln = lane & 15;
  int mbase = blockIdx.x * 256 + wid * 64;
  int nbase = blockIdx.y * 64;

  v8f acc[4][4];
#pragma unroll
  for (int mi = 0; mi < 4; ++mi)
#pragma unroll
    for (int ni = 0; ni < 4; ++ni) acc[mi][ni] = zero8();

  for (int kt = 0; kt < C; kt += 32) {
    Frag16 af[4], bf[4];
#pragma unroll
    for (int mi = 0; mi < 4; ++mi) {
      const unsigned short* arow = Ab + (size_t)(mbase + mi*16 + ln) * C + kt;
      __builtin_prefetch(arow + 32, 0, 1);
#pragma unroll
      for (int v = 0; v < 8; ++v)
        af[mi].u[v] = *(const unsigned int*)(arow + a_k0(v, hh));
    }
#pragma unroll
    for (int ni = 0; ni < 4; ++ni) {
      const unsigned short* brow = Wb + (size_t)(nbase + ni*16 + ln) * C + kt + 16*hh;
      __builtin_prefetch(brow + 32, 0, 1);
#pragma unroll
      for (int v = 0; v < 8; ++v)
        bf[ni].u[v] = *(const unsigned int*)(brow + 2*v);
    }
#pragma unroll
    for (int mi = 0; mi < 4; ++mi)
#pragma unroll
      for (int ni = 0; ni < 4; ++ni)
        acc[mi][ni] = __builtin_amdgcn_wmma_f32_16x16x32_bf16(
            false, af[mi].v, false, bf[ni].v, (short)0, acc[mi][ni], false, false);
  }

#pragma unroll
  for (int ni = 0; ni < 4; ++ni) {
    int ng = nbase + ni*16 + ln;
    float bi = bias[ng];
#pragma unroll
    for (int mi = 0; mi < 4; ++mi)
#pragma unroll
      for (int v = 0; v < 8; ++v) {
        int mg = mbase + mi*16 + v + 8*hh;
        out[(size_t)mg * C + ng] = acc[mi][ni][v] + bi;
      }
  }
}

extern "C" void kernel_launch(void* const* d_in, const int* in_sizes, int n_in,
                              void* d_out, int out_size, void* d_ws, size_t ws_size,
                              hipStream_t stream) {
  (void)in_sizes; (void)n_in; (void)out_size; (void)ws_size;
  const float*         x    = (const float*)d_in[0];
  const unsigned char* mask = (const unsigned char*)d_in[1];
  const float*         wqkv = (const float*)d_in[2];
  const float*         bqkv = (const float*)d_in[3];
  const float*         wo   = (const float*)d_in[4];
  const float*         bo   = (const float*)d_in[5];
  unsigned short* ws  = (unsigned short*)d_ws;
  float*          out = (float*)d_out;

  int nx = M_ROWS * C, nwq = N_QKV * C, nwo = C * C;
  cvt_bf16<<<(nx  + 255) / 256, 256, 0, stream>>>(x,    ws + XB_OFF, nx);
  cvt_bf16<<<(nwq + 255) / 256, 256, 0, stream>>>(wqkv, ws + WQ_OFF, nwq);
  cvt_bf16<<<(nwo + 255) / 256, 256, 0, stream>>>(wo,   ws + WO_OFF, nwo);

  qkv_gemm<<<dim3(M_ROWS / 256, N_QKV / 64), 128, 0, stream>>>(
      ws + XB_OFF, ws + WQ_OFF, bqkv, ws + Q_OFF, ws + K_OFF, ws + VT_OFF);

  flash_attn<<<(BB * NH * (S / 16)) / 4, 128, 0, stream>>>(
      ws + Q_OFF, ws + K_OFF, ws + VT_OFF, mask, ws + OC_OFF);

  proj_gemm<<<dim3(M_ROWS / 256, C / 64), 128, 0, stream>>>(
      ws + OC_OFF, ws + WO_OFF, bo, out);
}
